// CBERTProtoConcat_73504070304234
// MI455X (gfx1250) — matrix operand order repaired and checked by reading, not compiled, not run on hardware
//
#include <hip/hip_runtime.h>

// ---------------------------------------------------------------------------
// CDNA5 / gfx1250 implementation of the CBERT proto-concat scorer.
//   Hq = Qr @ W1[:, :D].T + b1      [128,1536]   (WMMA bf16 hi/lo split)
//   Hs = Sr @ W1[:, D:].T           [512,1536]   (WMMA bf16 hi/lo split)
//   scores[q,s] = sum_d relu(Hq[q,d] + Hs[s,d]) * W2[d] + b2   (VALU, LDS tiled)
//   agg[q,c]    = mean_{s: label==c} scores[q,s]
//   loss = -mean_q log_softmax(agg)[target_q];  preds == targets -> d_out[1..]
// ---------------------------------------------------------------------------

typedef __attribute__((ext_vector_type(16))) __bf16 v16bf;
typedef __attribute__((ext_vector_type(8)))  float  v8f;

#define DIM   768
#define TWOD  1536
#define QN    128
#define SN    512
#define CN    64

__device__ __forceinline__ float4 ldg4(const float* p) {
    return *reinterpret_cast<const float4*>(p);
}

// ---------------------------------------------------------------------------
// Kernel 1: unified hidden GEMM. Rows 0..127 -> query half (W1 cols [0,768)),
// rows 128..639 -> support half (W1 cols [768,1536)). One wave per 16x16 tile.
// C[m,n] = sum_k A[m,k] * W1[nColBase + n, koff + k],  k = 0..767
// ---------------------------------------------------------------------------
__global__ __launch_bounds__(32)
void hidden_gemm_wmma(const float* __restrict__ Qr,
                      const float* __restrict__ Sr,
                      const float* __restrict__ W1,
                      const float* __restrict__ b1,
                      float* __restrict__ Hq,
                      float* __restrict__ Hs)
{
    const int nt   = blockIdx.x;                 // 0..95  (col tile)
    const int mt   = blockIdx.y;                 // 0..39  (row tile)
    const int lane = threadIdx.x;                // 0..31
    const bool isQ = (mt < QN / 16);
    const float* A      = isQ ? Qr : Sr;
    const int   rowBase = (isQ ? mt : mt - QN / 16) * 16;
    const int   koff    = isQ ? 0 : DIM;

    const int mn    = lane & 15;
    const int khalf = (lane < 16) ? 0 : 8;       // unused (A-frag also uses kbase)
    const int kbase = (lane < 16) ? 0 : 16;      // B-frag K sub-offset

    const float* arow = A  + (size_t)(rowBase + mn) * DIM;
    const float* brow = W1 + (size_t)(nt * 16 + mn) * TWOD + koff;

    v8f acc = {0.f, 0.f, 0.f, 0.f, 0.f, 0.f, 0.f, 0.f};

    for (int k0 = 0; k0 < DIM; k0 += 32) {
        float a[16], b[16];
        float4 t;
        // A-frag: elems e -> K = kbase + e (contiguous, matches B-frag)
        t = ldg4(arow + k0 + kbase);           a[0]=t.x;  a[1]=t.y;  a[2]=t.z;  a[3]=t.w;
        t = ldg4(arow + k0 + kbase + 4);       a[4]=t.x;  a[5]=t.y;  a[6]=t.z;  a[7]=t.w;
        t = ldg4(arow + k0 + kbase + 8);       a[8]=t.x;  a[9]=t.y;  a[10]=t.z; a[11]=t.w;
        t = ldg4(arow + k0 + kbase + 12);      a[12]=t.x; a[13]=t.y; a[14]=t.z; a[15]=t.w;
        // B-frag: elems e -> K = kbase + e (contiguous along the W1 row)
        t = ldg4(brow + k0 + kbase);           b[0]=t.x;  b[1]=t.y;  b[2]=t.z;  b[3]=t.w;
        t = ldg4(brow + k0 + kbase + 4);       b[4]=t.x;  b[5]=t.y;  b[6]=t.z;  b[7]=t.w;
        t = ldg4(brow + k0 + kbase + 8);       b[8]=t.x;  b[9]=t.y;  b[10]=t.z; b[11]=t.w;
        t = ldg4(brow + k0 + kbase + 12);      b[12]=t.x; b[13]=t.y; b[14]=t.z; b[15]=t.w;

        // fp32 -> bf16 hi/lo split for near-fp32 accuracy on the tensor path
        v16bf ahi, alo, bhi, blo;
        #pragma unroll
        for (int e = 0; e < 16; ++e) {
            __bf16 h = (__bf16)a[e];
            ahi[e] = h;
            alo[e] = (__bf16)(a[e] - (float)h);
            __bf16 g = (__bf16)b[e];
            bhi[e] = g;
            blo[e] = (__bf16)(b[e] - (float)g);
        }
        acc = __builtin_amdgcn_wmma_f32_16x16x32_bf16(false, ahi, false, bhi,
                                                      (short)0, acc, false, false);
        acc = __builtin_amdgcn_wmma_f32_16x16x32_bf16(false, ahi, false, blo,
                                                      (short)0, acc, false, false);
        acc = __builtin_amdgcn_wmma_f32_16x16x32_bf16(false, alo, false, bhi,
                                                      (short)0, acc, false, false);
    }

    // C/D layout: VGPR v, lanes 0-15 -> M=v, lanes 16-31 -> M=v+8; N = lane%16
    const int   mOff = (lane < 16) ? 0 : 8;
    const int   col  = nt * 16 + mn;
    const float bias = isQ ? b1[col] : 0.0f;     // fold b1 into query rows once
    float* H = isQ ? Hq : Hs;
    #pragma unroll
    for (int v = 0; v < 8; ++v)
        H[(size_t)(rowBase + v + mOff) * TWOD + col] = acc[v] + bias;
}

// ---------------------------------------------------------------------------
// Kernel 2: fused relu-dot scores. 32x32 (q x s) tile per block, 256 threads,
// 2x2 register micro-tile, d staged through LDS in 32-wide chunks.
// ---------------------------------------------------------------------------
__global__ __launch_bounds__(256)
void score_tile(const float* __restrict__ Hq,
                const float* __restrict__ Hs,
                const float* __restrict__ W2,
                const float* __restrict__ b2,
                float* __restrict__ scores)
{
    __shared__ float lq[32][33];
    __shared__ float ls[32][33];
    __shared__ float lw[32];

    const int tid   = threadIdx.x;
    const int qBase = blockIdx.y * 32;
    const int sBase = blockIdx.x * 32;
    const int tq    = tid >> 4;    // 0..15 -> q rows 2*tq, 2*tq+1
    const int ts    = tid & 15;    // 0..15 -> s rows 2*ts, 2*ts+1

    float acc00 = 0.f, acc01 = 0.f, acc10 = 0.f, acc11 = 0.f;
    const float b2v = b2[0];

    for (int dc = 0; dc < TWOD; dc += 32) {
        #pragma unroll
        for (int r = 0; r < 4; ++r) {
            int lin  = r * 256 + tid;
            int row  = lin >> 5;
            int dcol = lin & 31;
            lq[row][dcol] = Hq[(size_t)(qBase + row) * TWOD + dc + dcol];
            ls[row][dcol] = Hs[(size_t)(sBase + row) * TWOD + dc + dcol];
        }
        if (tid < 32) lw[tid] = W2[dc + tid];
        __syncthreads();

        #pragma unroll 8
        for (int dd = 0; dd < 32; ++dd) {
            float w  = lw[dd];
            float a0 = lq[2 * tq][dd],     a1 = lq[2 * tq + 1][dd];
            float c0 = ls[2 * ts][dd],     c1 = ls[2 * ts + 1][dd];
            acc00 += fmaxf(a0 + c0, 0.f) * w;
            acc01 += fmaxf(a0 + c1, 0.f) * w;
            acc10 += fmaxf(a1 + c0, 0.f) * w;
            acc11 += fmaxf(a1 + c1, 0.f) * w;
        }
        __syncthreads();
    }

    const int q0 = qBase + 2 * tq;
    const int s0 = sBase + 2 * ts;
    scores[(size_t)q0 * SN + s0]           = acc00 + b2v;
    scores[(size_t)q0 * SN + s0 + 1]       = acc01 + b2v;
    scores[(size_t)(q0 + 1) * SN + s0]     = acc10 + b2v;
    scores[(size_t)(q0 + 1) * SN + s0 + 1] = acc11 + b2v;
}

// ---------------------------------------------------------------------------
// Kernel 3: per-query label-mean aggregation + log_softmax + NLL + pred check.
// One block (64 threads == 64 classes) per query.
// ---------------------------------------------------------------------------
__global__ __launch_bounds__(64)
void agg_softmax(const float* __restrict__ scores,
                 const int* __restrict__ labels,
                 const int* __restrict__ targets,
                 float* __restrict__ qloss,
                 float* __restrict__ out)      // out[1+q] = correctness flag
{
    const int q = blockIdx.x;
    const int c = threadIdx.x;                 // class id, 0..63
    __shared__ float sv[64];
    __shared__ float sa[64];
    __shared__ int   bidx;

    const float* row = scores + (size_t)q * SN;
    float sum = 0.f;
    int   cnt = 0;
    for (int s = 0; s < SN; ++s) {
        if (labels[s] == c) { sum += row[s]; ++cnt; }
    }
    const float agg = sum / fmaxf((float)cnt, 1.0f);

    sa[c] = agg;
    sv[c] = agg;
    if (c == 0) bidx = CN;
    __syncthreads();
    // max over classes
    for (int off = 32; off > 0; off >>= 1) {
        if (c < off) sv[c] = fmaxf(sv[c], sv[c + off]);
        __syncthreads();
    }
    const float m = sv[0];
    __syncthreads();
    // first-index argmax (matches jnp.argmax tie rule)
    if (agg == m) atomicMin(&bidx, c);
    sv[c] = expf(agg - m);
    __syncthreads();
    for (int off = 32; off > 0; off >>= 1) {
        if (c < off) sv[c] += sv[c + off];
        __syncthreads();
    }
    if (c == 0) {
        const int t = targets[q];
        const float logp = sa[t] - m - logf(sv[0]);
        qloss[q]   = -logp;
        out[1 + q] = (bidx == t) ? 1.0f : 0.0f;
    }
}

// ---------------------------------------------------------------------------
// Kernel 4: mean of per-query losses -> out[0]
// ---------------------------------------------------------------------------
__global__ __launch_bounds__(128)
void loss_reduce(const float* __restrict__ qloss, float* __restrict__ out)
{
    __shared__ float sv[128];
    const int t = threadIdx.x;
    sv[t] = qloss[t];
    __syncthreads();
    for (int off = 64; off > 0; off >>= 1) {
        if (t < off) sv[t] += sv[t + off];
        __syncthreads();
    }
    if (t == 0) out[0] = sv[0] / (float)QN;
}

// ---------------------------------------------------------------------------
extern "C" void kernel_launch(void* const* d_in, const int* in_sizes, int n_in,
                              void* d_out, int out_size, void* d_ws, size_t ws_size,
                              hipStream_t stream)
{
    const float* Qr      = (const float*)d_in[0];   // [128,768]
    const float* Sr      = (const float*)d_in[1];   // [512,768]
    const float* W1      = (const float*)d_in[2];   // [1536,1536]
    const float* b1      = (const float*)d_in[3];   // [1536]
    const float* W2      = (const float*)d_in[4];   // [1,1536]
    const float* b2      = (const float*)d_in[5];   // [1]
    const int*   labels  = (const int*)d_in[6];     // [512]
    const int*   targets = (const int*)d_in[7];     // [128]
    (void)in_sizes; (void)n_in; (void)out_size; (void)ws_size;

    float* Hq     = (float*)d_ws;                   // 128*1536
    float* Hs     = Hq + (size_t)QN * TWOD;         // 512*1536
    float* scores = Hs + (size_t)SN * TWOD;         // 128*512
    float* qloss  = scores + (size_t)QN * SN;       // 128
    float* out    = (float*)d_out;                  // [0]=loss, [1..128]=correct

    dim3 g1(TWOD / 16, (QN + SN) / 16);             // (96, 40)
    hidden_gemm_wmma<<<g1, 32, 0, stream>>>(Qr, Sr, W1, b1, Hq, Hs);

    dim3 g2(SN / 32, QN / 32);                      // (16, 4)
    score_tile<<<g2, 256, 0, stream>>>(Hq, Hs, W2, b2, scores);

    agg_softmax<<<QN, CN, 0, stream>>>(scores, labels, targets, qloss, out);

    loss_reduce<<<1, 128, 0, stream>>>(qloss, out);
}
